// AdvancedKANLayer_39934605918799
// MI455X (gfx1250) — matrix operand order repaired and checked by reading, not compiled, not run on hardware
//
#include <hip/hip_runtime.h>

// ---------------------------------------------------------------------------
// Fused KAN layer for MI455X (gfx1250, wave32).
//   out[M,N] = A[M, K] @ W[K, N],  K = 2048*9 (interleaved k = i*9 + g)
//     g=0: silu(LN(x)),  g=1..8: cos(pi*g*tanh(LN(x)))  (Chebyshev recurrence)
// pack weights (bf16) -> per-2048-row chunk { activations (bf16) -> WMMA GEMM }
// GEMM staging: GLOBAL_LOAD_ASYNC_TO_LDS_B128 + double-buffered LDS when the
// async builtins exist; otherwise a register-slim direct copy.
// ---------------------------------------------------------------------------

typedef __attribute__((ext_vector_type(8)))  float          v8f;
typedef __attribute__((ext_vector_type(16))) __bf16         v16bf;
typedef __attribute__((ext_vector_type(8)))  unsigned short v8us;
typedef __attribute__((ext_vector_type(16))) unsigned short v16us;
typedef __attribute__((ext_vector_type(4)))  int            v4i;

#define M_ROWS  16384
#define D_IN    2048
#define D_OUT   2048
#define G_FREQ  8
#define K_TOT   (9 * D_IN)     // 18432
#define MCHUNK  2048

#define BM 128
#define BN 128
#define BK 64
#define KP 72                  // padded LDS row stride (ushort) -> conflict-free

#if defined(__has_builtin)
#if __has_builtin(__builtin_amdgcn_global_load_async_to_lds_b128)
#define USE_ASYNC_COPY 1
#endif
#endif

#ifdef USE_ASYNC_COPY
typedef __attribute__((address_space(1))) v4i gv4i;   // global int4
typedef __attribute__((address_space(3))) v4i lv4i;   // LDS int4
static __device__ __forceinline__ void wait_async0() {
#if __has_builtin(__builtin_amdgcn_s_wait_asynccnt)
    __builtin_amdgcn_s_wait_asynccnt(0);
#else
    asm volatile("s_wait_asynccnt 0x0" ::: "memory");
#endif
}
#endif

static __device__ __forceinline__ unsigned short f2bf(float f) {
    unsigned u = __builtin_bit_cast(unsigned, f);
    u += 0x7FFFu + ((u >> 16) & 1u);
    return (unsigned short)(u >> 16);
}

// ---------------- Kernel 1: pack weights -> bf16 Wb[n][k] ------------------
__global__ __launch_bounds__(256) void pack_w_kernel(
    const float* __restrict__ bw, const float* __restrict__ sw,
    unsigned short* __restrict__ Wb)
{
    const int n   = blockIdx.x;
    const int tid = threadIdx.x;
    __align__(16) unsigned short outv[72];
#pragma unroll
    for (int q = 0; q < 8; ++q) {
        const size_t base = (size_t)n * D_IN + (tid * 8 + q);
        const float b = bw[base];
        const float4* sp = (const float4*)(sw + base * G_FREQ);
        const float4 s0 = sp[0], s1 = sp[1];
        outv[q * 9 + 0] = f2bf(b);
        outv[q * 9 + 1] = f2bf(s0.x); outv[q * 9 + 2] = f2bf(s0.y);
        outv[q * 9 + 3] = f2bf(s0.z); outv[q * 9 + 4] = f2bf(s0.w);
        outv[q * 9 + 5] = f2bf(s1.x); outv[q * 9 + 6] = f2bf(s1.y);
        outv[q * 9 + 7] = f2bf(s1.z); outv[q * 9 + 8] = f2bf(s1.w);
    }
    uint4* dst = (uint4*)(Wb + (size_t)n * K_TOT + tid * 72);
#pragma unroll
    for (int w = 0; w < 9; ++w) dst[w] = ((const uint4*)outv)[w];
}

// ---------------- Kernel 2: LN + activations -> bf16 Aact[row][k] ----------
__global__ __launch_bounds__(256) void act_kernel(
    const float* __restrict__ x,
    const float* __restrict__ lnw, const float* __restrict__ lnb,
    unsigned short* __restrict__ Aact, int row_base)
{
    const int row = row_base + blockIdx.x;
    const int tid = threadIdx.x;
    __shared__ float xs[D_IN];
    __shared__ float red0[256], red1[256];
    __shared__ float smean, srstd;

    float s = 0.f, s2 = 0.f;
#pragma unroll
    for (int m = 0; m < 8; ++m) {
        float v = x[(size_t)row * D_IN + tid + 256 * m];
        xs[tid + 256 * m] = v;
        s += v; s2 += v * v;
    }
    red0[tid] = s; red1[tid] = s2;
    __syncthreads();
    for (int off = 128; off > 0; off >>= 1) {
        if (tid < off) { red0[tid] += red0[tid + off]; red1[tid] += red1[tid + off]; }
        __syncthreads();
    }
    if (tid == 0) {
        float m = red0[0] * (1.0f / D_IN);
        float v = red1[0] * (1.0f / D_IN) - m * m;
        smean = m;
        srstd = __frsqrt_rn(v + 1e-5f);
    }
    __syncthreads();
    const float mean = smean, rs = srstd;
    const float PI = 3.14159265358979323846f;

    __align__(16) unsigned short outv[72];
#pragma unroll
    for (int q = 0; q < 8; ++q) {
        const int i = tid * 8 + q;
        const float xn = (xs[i] - mean) * rs * lnw[i] + lnb[i];
        const float sig = __frcp_rn(1.0f + __expf(-xn));
        outv[q * 9 + 0] = f2bf(xn * sig);
        const float e2 = __expf(xn + xn);
        const float t  = __builtin_fmaf(-2.0f, __frcp_rn(e2 + 1.0f), 1.0f);
        const float c1 = __cosf(PI * t);
        const float twoc1 = c1 + c1;
        float cm2 = 1.0f, cm1 = c1;
        outv[q * 9 + 1] = f2bf(c1);
#pragma unroll
        for (int g = 2; g <= 8; ++g) {
            const float c = __builtin_fmaf(twoc1, cm1, -cm2);
            outv[q * 9 + g] = f2bf(c);
            cm2 = cm1; cm1 = c;
        }
    }
    uint4* dst = (uint4*)(Aact + (size_t)row * K_TOT + tid * 72);
#pragma unroll
    for (int w = 0; w < 9; ++w) dst[w] = ((const uint4*)outv)[w];
}

// ---------------- Kernel 3: bf16 WMMA GEMM -------------------------------
__global__ __launch_bounds__(256) void kan_gemm_kernel(
    const unsigned short* __restrict__ Aact,   // [MCHUNK][K_TOT] bf16
    const unsigned short* __restrict__ Wb,     // [D_OUT][K_TOT] bf16
    float* __restrict__ out, int row_base)
{
#ifdef USE_ASYNC_COPY
    __shared__ __align__(16) unsigned short As[2][BM * KP];
    __shared__ __align__(16) unsigned short Bs[2][BN * KP];
#else
    __shared__ __align__(16) unsigned short As[1][BM * KP];
    __shared__ __align__(16) unsigned short Bs[1][BN * KP];
#endif

    const int tid  = threadIdx.x;
    const int row0 = blockIdx.x * BM;
    const int col0 = blockIdx.y * BN;
    const int lane = tid & 31;
    const int wave = tid >> 5;
    const int wm   = wave & 1;          // 2 waves x 64 rows
    const int wn   = wave >> 1;         // 4 waves x 32 cols
    const int l    = lane & 15;
    const int hi   = lane >> 4;

    // copy mapping: this thread handles rows {rrow + 32*j}, k-slot ccol (16B)
    const int rrow = tid >> 3;          // 0..31
    const int ccol = (tid & 7) * 8;     // ushort offset, 16B aligned
    const int lbase = rrow * KP + ccol; // LDS slot (j adds 32*KP)
    const size_t gbaseA = (size_t)(row0 + rrow) * K_TOT + ccol;
    const size_t gbaseB = (size_t)(col0 + rrow) * K_TOT + ccol;

    v8f acc[4][2];
#pragma unroll
    for (int i = 0; i < 4; ++i)
#pragma unroll
        for (int j = 0; j < 2; ++j)
            acc[i][j] = (v8f)0.0f;

    const int NT = K_TOT / BK;          // 288

#ifdef USE_ASYNC_COPY
    // async DMA of one K-tile into LDS buffer b (8 x b128 per thread, no VGPR data)
    auto issue_tile = [&](int kt, int b) {
#pragma unroll
        for (int j = 0; j < 4; ++j) {
            __builtin_amdgcn_global_load_async_to_lds_b128(
                (gv4i*)(Aact + gbaseA + (size_t)kt * BK + (size_t)j * (32 * K_TOT)),
                (lv4i*)&As[b][lbase + j * (32 * KP)], 0, 0);
            __builtin_amdgcn_global_load_async_to_lds_b128(
                (gv4i*)(Wb + gbaseB + (size_t)kt * BK + (size_t)j * (32 * K_TOT)),
                (lv4i*)&Bs[b][lbase + j * (32 * KP)], 0, 0);
        }
    };
    issue_tile(0, 0);
#endif

    for (int kt = 0; kt < NT; ++kt) {
#ifdef USE_ASYNC_COPY
        const int b = kt & 1;
        wait_async0();                  // this wave's DMA into buf b complete
        __syncthreads();                // all waves' DMA complete; prev compute done
        if (kt + 1 < NT) issue_tile(kt + 1, b ^ 1);  // overlaps WMMA below
#else
        const int b = 0;
        uint4 va[4], vb[4];
#pragma unroll
        for (int j = 0; j < 4; ++j) {
            va[j] = *(const uint4*)(Aact + gbaseA + (size_t)kt * BK + (size_t)j * (32 * K_TOT));
            vb[j] = *(const uint4*)(Wb   + gbaseB + (size_t)kt * BK + (size_t)j * (32 * K_TOT));
        }
        __syncthreads();
#pragma unroll
        for (int j = 0; j < 4; ++j) {
            *(uint4*)&As[0][lbase + j * (32 * KP)] = va[j];
            *(uint4*)&Bs[0][lbase + j * (32 * KP)] = vb[j];
        }
        __syncthreads();
#endif

#pragma unroll
        for (int ak = 0; ak < BK; ak += 32) {
            v16bf afrag[4];
            v16bf bfrag[2];
#pragma unroll
            for (int mt = 0; mt < 4; ++mt) {
                const int o0 = (wm * 64 + mt * 16 + l) * KP + ak + hi * 8;
                v8us lo  = *(const v8us*)&As[b][o0];
                v8us hi8 = *(const v8us*)&As[b][o0 + 16];
                v16us c = __builtin_shufflevector(lo, hi8,
                    0,1,2,3,4,5,6,7,8,9,10,11,12,13,14,15);
                afrag[mt] = __builtin_bit_cast(v16bf, c);
            }
#pragma unroll
            for (int nt = 0; nt < 2; ++nt) {
                const int o0 = (wn * 32 + nt * 16 + l) * KP + ak + hi * 8;
                v8us lo  = *(const v8us*)&Bs[b][o0];
                v8us hi8 = *(const v8us*)&Bs[b][o0 + 16];
                v16us c = __builtin_shufflevector(lo, hi8,
                    0,1,2,3,4,5,6,7,8,9,10,11,12,13,14,15);
                bfrag[nt] = __builtin_bit_cast(v16bf, c);
            }
#pragma unroll
            for (int mt = 0; mt < 4; ++mt)
#pragma unroll
                for (int nt = 0; nt < 2; ++nt)
                    acc[mt][nt] = __builtin_amdgcn_wmma_f32_16x16x32_bf16(
                        false, afrag[mt], false, bfrag[nt],
                        (short)0, acc[mt][nt], false, false);
        }
    }

    // epilogue: C layout VGPR r -> (M = hi*8 + r, N = l)
#pragma unroll
    for (int mt = 0; mt < 4; ++mt) {
#pragma unroll
        for (int nt = 0; nt < 2; ++nt) {
            const int m0 = row_base + row0 + wm * 64 + mt * 16 + hi * 8;
            const int n0 = col0 + wn * 32 + nt * 16 + l;
#pragma unroll
            for (int r = 0; r < 8; ++r)
                out[(size_t)(m0 + r) * D_OUT + n0] = acc[mt][nt][r];
        }
    }
}

// ---------------------------------------------------------------------------
extern "C" void kernel_launch(void* const* d_in, const int* in_sizes, int n_in,
                              void* d_out, int out_size, void* d_ws, size_t ws_size,
                              hipStream_t stream) {
    const float* x   = (const float*)d_in[0];
    const float* lnw = (const float*)d_in[1];
    const float* lnb = (const float*)d_in[2];
    const float* bw  = (const float*)d_in[3];
    const float* sw  = (const float*)d_in[4];
    float* out = (float*)d_out;

    unsigned short* Wb   = (unsigned short*)d_ws;            // [2048][18432] bf16
    unsigned short* Aact = Wb + (size_t)D_OUT * K_TOT;       // [2048][18432] bf16

    pack_w_kernel<<<D_OUT, 256, 0, stream>>>(bw, sw, Wb);

    const dim3 ggrid(MCHUNK / BM, D_OUT / BN);               // (16, 16)
    for (int c = 0; c < M_ROWS / MCHUNK; ++c) {              // 8 chunks
        const int row_base = c * MCHUNK;
        act_kernel<<<MCHUNK, 256, 0, stream>>>(x, lnw, lnb, Aact, row_base);
        kan_gemm_kernel<<<ggrid, 256, 0, stream>>>(Aact, Wb, out, row_base);
    }
}
